// HdcMamba9v3Block_60052232732975
// MI455X (gfx1250) — compile-verified, matrix operand order checked by reading, not run on hardware
//
#include <hip/hip_runtime.h>

typedef __attribute__((ext_vector_type(16))) _Float16 v16h;
typedef __attribute__((ext_vector_type(8)))  float    v8f;
typedef int v4i_ __attribute__((vector_size(16)));

union Frag { v16h h; unsigned int u[8]; };

static __device__ __forceinline__ unsigned int pack2(float a, float b) {
  union { _Float16 h[2]; unsigned int u; } p;
  p.h[0] = (_Float16)a; p.h[1] = (_Float16)b; return p.u;
}
static __device__ __forceinline__ float relu2f(float t) { return t > 0.f ? t * t : 0.f; }

// ---- gfx1250 async global->LDS path (guarded; falls back to sync copy) ----
#if defined(__has_builtin)
# if __has_builtin(__builtin_amdgcn_global_load_async_to_lds_b128)
#  define HAVE_ASYNC_LDS 1
# endif
#endif
#ifndef HAVE_ASYNC_LDS
# define HAVE_ASYNC_LDS 0
#endif

#define AS1 __attribute__((address_space(1)))
#define AS3 __attribute__((address_space(3)))

static __device__ __forceinline__ void async_copy_b128(const _Float16* g, _Float16* l) {
#if HAVE_ASYNC_LDS
  __builtin_amdgcn_global_load_async_to_lds_b128(
      (AS1 v4i_*)(unsigned long long)(const void*)g,
      (AS3 v4i_*)(void*)l, 0, 0);
#else
  *(uint4*)l = *(const uint4*)g;
#endif
}
static __device__ __forceinline__ void wait_async0() {
#if HAVE_ASYNC_LDS
# if __has_builtin(__builtin_amdgcn_s_wait_asynccnt)
  __builtin_amdgcn_s_wait_asynccnt(0);
# else
  asm volatile("s_wait_asynccnt 0x0" ::: "memory");
# endif
#endif
}

// ---------------------------------------------------------------- fp32 -> fp16
__global__ __launch_bounds__(256) void f32_to_f16(const float* __restrict__ in,
                                                  _Float16* __restrict__ out, int n) {
  int i = blockIdx.x * 256 + threadIdx.x;
  if (i < n) out[i] = (_Float16)in[i];
}

// ---------------------------------------------------------------- LayerNorm over D=1024
__global__ __launch_bounds__(256) void ln_kernel(const float* __restrict__ x,
                                                 const float* __restrict__ w,
                                                 const float* __restrict__ bb,
                                                 float* __restrict__ out) {
  int row = blockIdx.x, tid = threadIdx.x;
  float4 xv = *((const float4*)(x + (size_t)row * 1024) + tid);
  float s  = xv.x + xv.y + xv.z + xv.w;
  float s2 = xv.x * xv.x + xv.y * xv.y + xv.z * xv.z + xv.w * xv.w;
  __shared__ float r1[256], r2[256];
  r1[tid] = s; r2[tid] = s2;
  __syncthreads();
  for (int st = 128; st > 0; st >>= 1) {
    if (tid < st) { r1[tid] += r1[tid + st]; r2[tid] += r2[tid + st]; }
    __syncthreads();
  }
  float mu = r1[0] * (1.f / 1024.f);
  float var = r2[0] * (1.f / 1024.f) - mu * mu;
  float rs = rsqrtf(var + 1e-5f);
  int c = tid * 4;
  float4 o;
  o.x = (xv.x - mu) * rs * w[c + 0] + bb[c + 0];
  o.y = (xv.y - mu) * rs * w[c + 1] + bb[c + 1];
  o.z = (xv.z - mu) * rs * w[c + 2] + bb[c + 2];
  o.w = (xv.w - mu) * rs * w[c + 3] + bb[c + 3];
  *((float4*)(out + (size_t)row * 1024) + tid) = o;
}

// ---------------------------------------------------------------- causal depthwise conv (K=4) + SiLU
__global__ __launch_bounds__(256) void conv_silu(const float* __restrict__ hln,
                                                 const float* __restrict__ cw,
                                                 const float* __restrict__ cb,
                                                 _Float16* __restrict__ h16) {
  int idx = blockIdx.x * 256 + threadIdx.x;
  int d = idx & 1023, l = (idx >> 10) & 4095, b = idx >> 22;
  float acc = cb[d];
#pragma unroll
  for (int k = 0; k < 4; ++k) {
    int ls = l - 3 + k;
    if (ls >= 0) acc = fmaf(hln[(((size_t)b << 12) + ls) * 1024 + d], cw[d * 4 + k], acc);
  }
  float sig = 1.f / (1.f + __expf(-acc));
  h16[idx] = (_Float16)(acc * sig);
}

// ---------------------------------------------------------------- WMMA GEMM  C = A(f16,MxK) @ B(f16,KxN) + bias (+res)
// Double-buffered LDS; A tile staged via GLOBAL_LOAD_ASYNC_TO_LDS_B128 (ASYNCcnt),
// B tile transposed into LDS [n][k] for the 16-bit B fragment layout.
template <bool OUT_F16, bool HAS_RES>
__global__ __launch_bounds__(256)
void gemm_wmma(const _Float16* __restrict__ A, const _Float16* __restrict__ Bm,
               const float* __restrict__ bias, const float* __restrict__ res,
               float* __restrict__ outF, _Float16* __restrict__ outH,
               int M, int N, int K, int ldOut) {
  __shared__ _Float16 As[2][128 * 40];
  __shared__ _Float16 Bs[2][64 * 40];
  const int tid = threadIdx.x, lane = tid & 31, wv = tid >> 5;
  const int wm = wv >> 1, wn = wv & 1;
  const int bm = blockIdx.x * 128, bn = blockIdx.y * 64;
  const int hi = (lane >= 16) ? 8 : 0;

  auto stageA = [&](int buf, int k0) {
#pragma unroll
    for (int u0 = 0; u0 < 512; u0 += 256) {
      int u = u0 + tid;
      int row = u >> 2, c4 = u & 3;
      async_copy_b128(A + (size_t)(bm + row) * K + k0 + c4 * 8, &As[buf][row * 40 + c4 * 8]);
    }
  };
  auto stageB = [&](int buf, int k0) {
    int kr = tid >> 3, n8 = (tid & 7) * 8;
    uint4 val = *(const uint4*)(Bm + (size_t)(k0 + kr) * N + bn + n8);
    const _Float16* hp = (const _Float16*)&val;
#pragma unroll
    for (int j = 0; j < 8; ++j) Bs[buf][(n8 + j) * 40 + kr] = hp[j];
  };

  v8f acc[2][2] = {};
  stageA(0, 0);
  stageB(0, 0);
  for (int k0 = 0; k0 < K; k0 += 32) {
    int buf = (k0 >> 5) & 1;
    wait_async0();          // own ASYNCcnt drained -> LDS writes of this wave done
    __syncthreads();        // all waves' tile writes visible; prior reads of buf^1 done
    if (k0 + 32 < K) {      // prefetch next tile into the other buffer (overlaps compute)
      stageA(buf ^ 1, k0 + 32);
      stageB(buf ^ 1, k0 + 32);
    }
    Frag af[2], bf[2];
#pragma unroll
    for (int i = 0; i < 2; ++i) {
      const _Float16* pa = &As[buf][(wm * 32 + i * 16 + (lane & 15)) * 40 + hi];
      const _Float16* pb = &Bs[buf][(wn * 32 + i * 16 + (lane & 15)) * 40 + hi];
#pragma unroll
      for (int j = 0; j < 8; ++j) {
        int kb = (j < 4) ? 2 * j : 8 + 2 * j;   // 16-bit A/B fragment K pattern
        af[i].u[j] = *(const unsigned int*)(pa + kb);
        bf[i].u[j] = *(const unsigned int*)(pb + kb);
      }
    }
#pragma unroll
    for (int i = 0; i < 2; ++i)
#pragma unroll
      for (int j = 0; j < 2; ++j)
        acc[i][j] = __builtin_amdgcn_wmma_f32_16x16x32_f16(false, af[i].h, false, bf[j].h,
                                                           (short)0, acc[i][j], false, false);
  }
#pragma unroll
  for (int i = 0; i < 2; ++i)
#pragma unroll
    for (int j = 0; j < 2; ++j)
#pragma unroll
      for (int r = 0; r < 8; ++r) {
        int grow = bm + wm * 32 + i * 16 + r + hi;
        int gcol = bn + wn * 32 + j * 16 + (lane & 15);
        float v = acc[i][j][r] + bias[gcol];
        if (HAS_RES) v += res[(size_t)grow * N + gcol];
        if (OUT_F16) outH[(size_t)grow * ldOut + gcol] = (_Float16)v;
        else         outF[(size_t)grow * ldOut + gcol] = v;
      }
}

// ---------------------------------------------------------------- SSM affine scan, one wave per (b,s)
__global__ __launch_bounds__(32) void ssm_scan(const _Float16* __restrict__ projs,
                                               const float* __restrict__ A_log,
                                               _Float16* __restrict__ concat) {
  int bs = blockIdx.x, b = bs >> 7, s = bs & 127, lane = threadIdx.x;
  float eA = __expf(A_log[s]);
  float carry = 0.f;
  for (int l0 = 0; l0 < 4096; l0 += 32) {
    int l = l0 + lane;
    size_t rb = ((size_t)(b * 4096 + l)) * 4480;
    float dtr = (float)projs[rb + s];
    float xs  = (float)projs[rb + 128 + s];
    float bss = (float)projs[rb + 256 + s];
    float dt = dtr > 20.f ? dtr : log1pf(__expf(dtr));
    float a = __expf(-dt * eA);
    float u = dt * bss * xs;
#pragma unroll
    for (int off = 1; off < 32; off <<= 1) {
      float ap = __shfl_up(a, off, 32);
      float up = __shfl_up(u, off, 32);
      if (lane >= off) { u = fmaf(a, up, u); a *= ap; }
    }
    float y = fmaf(a, carry, u);
    concat[((size_t)(b * 4096 + l)) * 1152 + s] = (_Float16)y;
    carry = __shfl(y, 31, 32);
  }
}

// ---------------------------------------------------------------- intra-chunk attention + k^T v  (per b,h,n)
__global__ __launch_bounds__(256)
void attn_intra(const _Float16* __restrict__ projs, const float* __restrict__ gamma,
                float* __restrict__ yintra, float* __restrict__ kvbuf) {
  __shared__ _Float16 sm[25088];
  _Float16* vT  = sm;            // 64 x 130   v transposed [d][c]
  _Float16* ks  = sm + 8320;     // 128 x 66   relu2(k) [c][d]
  _Float16* ksT = sm + 16768;    // 64 x 130   (relu2(k)*kdec)^T [d][c]
  _Float16* sc  = sm + 8320;     // 128 x 130  scores (overlays ks+ksT after use)

  const int n0 = blockIdx.x, h = blockIdx.y, b = blockIdx.z;
  const int tid = threadIdx.x, lane = tid & 31, w = tid >> 5;
  const int hi = (lane >= 16) ? 8 : 0;
  const float lg = logf(gamma[h]);
  const int qc0 = 384 + h * 64, kc0 = 1408 + h * 64, vc0 = 2432 + h * 64;
  const size_t rowB = (size_t)(b * 4096 + n0 * 128);

  for (int i = tid; i < 8192; i += 256) {
    int c = i >> 6, d = i & 63;
    size_t rp = (rowB + c) * 4480;
    float kf = relu2f((float)projs[rp + kc0 + d]);
    ks[c * 66 + d]   = (_Float16)kf;
    ksT[d * 130 + c] = (_Float16)(kf * __expf((127 - c) * lg));
    vT[d * 130 + c]  = projs[rp + vc0 + d];
  }
  __syncthreads();

  // ---- kv = ksT(64x128) @ v(128x64); each wave owns 2 of the 16 tiles
  {
    int tr = w >> 1;
    size_t base = (((size_t)(b * 16 + h)) * 32 + n0) * 4096;
    for (int t = 0; t < 2; ++t) {
      int tc = (w & 1) * 2 + t;
      v8f acc = {};
      for (int kk = 0; kk < 4; ++kk) {
        Frag a, bf;
        const _Float16* pa = ksT + (tr * 16 + (lane & 15)) * 130 + kk * 32 + hi;
        const _Float16* pb = vT  + (tc * 16 + (lane & 15)) * 130 + kk * 32 + hi;
#pragma unroll
        for (int j = 0; j < 8; ++j) {
          int kb = (j < 4) ? 2 * j : 8 + 2 * j;
          a.u[j]  = *(const unsigned int*)(pa + kb);
          bf.u[j] = *(const unsigned int*)(pb + kb);
        }
        acc = __builtin_amdgcn_wmma_f32_16x16x32_f16(false, a.h, false, bf.h, (short)0, acc, false, false);
      }
#pragma unroll
      for (int r = 0; r < 8; ++r)
        kvbuf[base + (tr * 16 + r + hi) * 64 + tc * 16 + (lane & 15)] = acc[r];
    }
  }

  // ---- scores strip (rows 16w..16w+15): A = relu2(q) fragments from global, B = ks
  Frag aq[2];
  {
    int c = 16 * w + (lane & 15);
    size_t rp = (rowB + c) * 4480 + qc0;
#pragma unroll
    for (int kk = 0; kk < 2; ++kk)
#pragma unroll
      for (int j = 0; j < 8; ++j) {
        int kb = ((j < 4) ? 2 * j : 8 + 2 * j) + hi + kk * 32;
        float q0 = relu2f((float)projs[rp + kb]);
        float q1 = relu2f((float)projs[rp + kb + 1]);
        aq[kk].u[j] = pack2(q0, q1);
      }
  }
  float svals[8][8];
#pragma unroll
  for (int j = 0; j < 8; ++j) {
    v8f acc = {};
#pragma unroll
    for (int kk = 0; kk < 2; ++kk) {
      Frag bf;
      const _Float16* pb = ks + (j * 16 + (lane & 15)) * 66 + kk * 32 + hi;
#pragma unroll
      for (int jj = 0; jj < 8; ++jj) {
        int kb = (jj < 4) ? 2 * jj : 8 + 2 * jj;
        bf.u[jj] = *(const unsigned int*)(pb + kb);
      }
      acc = __builtin_amdgcn_wmma_f32_16x16x32_f16(false, aq[kk].h, false, bf.h, (short)0, acc, false, false);
    }
#pragma unroll
    for (int r = 0; r < 8; ++r) {
      int c = 16 * w + r + hi, m = j * 16 + (lane & 15);
      svals[j][r] = (m <= c) ? acc[r] * __expf((c - m) * lg) : 0.f;
    }
  }
  __syncthreads();   // all waves done reading ks/ksT before overlay
#pragma unroll
  for (int j = 0; j < 8; ++j)
#pragma unroll
    for (int r = 0; r < 8; ++r)
      sc[(16 * w + r + hi) * 130 + j * 16 + (lane & 15)] = (_Float16)svals[j][r];
  __syncthreads();

  // ---- y_intra = scores(128x128) @ v(128x64)
  {
    size_t base = (((size_t)(b * 16 + h)) * 32 + n0) * 8192;
#pragma unroll
    for (int jt = 0; jt < 4; ++jt) {
      v8f acc = {};
      for (int kk = 0; kk < 4; ++kk) {
        Frag a, bf;
        const _Float16* pa = sc + (16 * w + (lane & 15)) * 130 + kk * 32 + hi;
        const _Float16* pb = vT + (jt * 16 + (lane & 15)) * 130 + kk * 32 + hi;
#pragma unroll
        for (int j = 0; j < 8; ++j) {
          int kb = (j < 4) ? 2 * j : 8 + 2 * j;
          a.u[j]  = *(const unsigned int*)(pa + kb);
          bf.u[j] = *(const unsigned int*)(pb + kb);
        }
        acc = __builtin_amdgcn_wmma_f32_16x16x32_f16(false, a.h, false, bf.h, (short)0, acc, false, false);
      }
#pragma unroll
      for (int r = 0; r < 8; ++r)
        yintra[base + (16 * w + r + hi) * 64 + jt * 16 + (lane & 15)] = acc[r];
    }
  }
}

// ---------------------------------------------------------------- sequential inter-chunk state recurrence
__global__ __launch_bounds__(256) void chunk_states(const float* __restrict__ kvbuf,
                                                    const float* __restrict__ gamma,
                                                    _Float16* __restrict__ statesH) {
  int bh = blockIdx.x, h = bh & 15, t = threadIdx.x;
  float gC = __expf(128.f * logf(gamma[h]));
  float st[16];
#pragma unroll
  for (int i = 0; i < 16; ++i) st[i] = 0.f;
  size_t base = (size_t)bh * 32 * 4096;
  for (int n = 0; n < 32; ++n) {
    size_t off = base + (size_t)n * 4096 + t * 16;
#pragma unroll
    for (int i = 0; i < 16; ++i) {
      statesH[off + i] = (_Float16)st[i];      // pre-update state feeds chunk n
      st[i] = st[i] * gC + kvbuf[off + i];
    }
  }
}

// ---------------------------------------------------------------- y_inter + groupnorm + gate  (per b,h,n)
__global__ __launch_bounds__(256)
void attn_final(const _Float16* __restrict__ projs, const _Float16* __restrict__ statesH,
                const float* __restrict__ yintra, const float* __restrict__ gamma,
                const float* __restrict__ gn_w, const float* __restrict__ gn_b,
                _Float16* __restrict__ concat) {
  __shared__ _Float16 sm[12672];
  _Float16* qds = sm;         // 128 x 66 : relu2(q)*gamma^c
  _Float16* stT = sm + 8448;  // 64 x 66  : state transposed [e][d]

  const int n0 = blockIdx.x, h = blockIdx.y, b = blockIdx.z;
  const int tid = threadIdx.x, lane = tid & 31, w = tid >> 5;
  const int hi = (lane >= 16) ? 8 : 0;
  const float lg = logf(gamma[h]);
  const int qc0 = 384 + h * 64, gc0 = 3456 + h * 64;
  const size_t rowB = (size_t)(b * 4096 + n0 * 128);
  const size_t sbase = (((size_t)(b * 16 + h)) * 32 + n0) * 4096;
  const size_t ybase = (((size_t)(b * 16 + h)) * 32 + n0) * 8192;

  for (int i = tid; i < 8192; i += 256) {
    int c = i >> 6, d = i & 63;
    float qf = relu2f((float)projs[(rowB + c) * 4480 + qc0 + d]);
    qds[c * 66 + d] = (_Float16)(qf * __expf(c * lg));
  }
  for (int i = tid; i < 4096; i += 256) {
    int d = i >> 6, e = i & 63;
    stT[e * 66 + d] = statesH[sbase + i];
  }
  __syncthreads();

  float vals[4][8];
#pragma unroll
  for (int jt = 0; jt < 4; ++jt) {
    v8f acc = {};
#pragma unroll
    for (int kk = 0; kk < 2; ++kk) {
      Frag a, bf;
      const _Float16* pa = qds + (16 * w + (lane & 15)) * 66 + kk * 32 + hi;
      const _Float16* pb = stT + (jt * 16 + (lane & 15)) * 66 + kk * 32 + hi;
#pragma unroll
      for (int j = 0; j < 8; ++j) {
        int kb = (j < 4) ? 2 * j : 8 + 2 * j;
        a.u[j]  = *(const unsigned int*)(pa + kb);
        bf.u[j] = *(const unsigned int*)(pb + kb);
      }
      acc = __builtin_amdgcn_wmma_f32_16x16x32_f16(false, a.h, false, bf.h, (short)0, acc, false, false);
    }
#pragma unroll
    for (int r = 0; r < 8; ++r)
      vals[jt][r] = acc[r] + yintra[ybase + (16 * w + r + hi) * 64 + jt * 16 + (lane & 15)];
  }
  // groupnorm over Dh=64 per row, then sigmoid(gate)
#pragma unroll
  for (int r = 0; r < 8; ++r) {
    float s = 0.f, s2 = 0.f;
#pragma unroll
    for (int jt = 0; jt < 4; ++jt) { s += vals[jt][r]; s2 += vals[jt][r] * vals[jt][r]; }
#pragma unroll
    for (int m = 8; m >= 1; m >>= 1) {
      s  += __shfl_xor(s, m, 32);
      s2 += __shfl_xor(s2, m, 32);
    }
    float mu = s * (1.f / 64.f);
    float var = s2 * (1.f / 64.f) - mu * mu;
    float rs = rsqrtf(var + 1e-5f);
    int c = 16 * w + r + hi;
    size_t rp = (rowB + c) * 4480;
    size_t cp = (rowB + c) * 1152;
#pragma unroll
    for (int jt = 0; jt < 4; ++jt) {
      int e = jt * 16 + (lane & 15);
      float g = (float)projs[rp + gc0 + e];
      float sig = 1.f / (1.f + __expf(-g));
      float yv = ((vals[jt][r] - mu) * rs * gn_w[h * 64 + e] + gn_b[h * 64 + e]) * sig;
      concat[cp + 128 + h * 64 + e] = (_Float16)yv;
    }
  }
}

// ----------------------------------------------------------------
extern "C" void kernel_launch(void* const* d_in, const int* in_sizes, int n_in,
                              void* d_out, int out_size, void* d_ws, size_t ws_size,
                              hipStream_t stream) {
  (void)in_sizes; (void)n_in; (void)out_size; (void)ws_size;
  const float* x      = (const float*)d_in[0];
  const float* ln_w   = (const float*)d_in[1];
  const float* ln_b   = (const float*)d_in[2];
  const float* conv_w = (const float*)d_in[3];
  const float* conv_b = (const float*)d_in[4];
  const float* W_in   = (const float*)d_in[5];
  const float* b_in   = (const float*)d_in[6];
  const float* A_log  = (const float*)d_in[7];
  const float* gamma  = (const float*)d_in[8];
  const float* gn_w   = (const float*)d_in[9];
  const float* gn_b   = (const float*)d_in[10];
  const float* W_out  = (const float*)d_in[11];
  const float* b_out  = (const float*)d_in[12];
  float* out = (float*)d_out;
  char* ws = (char*)d_ws;

  size_t o = 0;
  auto alloc = [&](size_t bytes) { size_t r = o; o += (bytes + 255) & ~(size_t)255; return r; };
  float*    hln     = (float*)   (ws + alloc((size_t)16384 * 1024 * 4));
  _Float16* h16     = (_Float16*)(ws + alloc((size_t)16384 * 1024 * 2));
  _Float16* Wi16    = (_Float16*)(ws + alloc((size_t)1024 * 4480 * 2));
  _Float16* projs   = (_Float16*)(ws + alloc((size_t)16384 * 4480 * 2));
  _Float16* concat  = (_Float16*)(ws + alloc((size_t)16384 * 1152 * 2));
  float*    yintra  = (float*)   (ws + alloc((size_t)16384 * 1024 * 4));
  float*    kvbuf   = (float*)   (ws + alloc((size_t)2048 * 4096 * 4));
  _Float16* statesH = (_Float16*)(ws + alloc((size_t)2048 * 4096 * 2));
  _Float16* Wo16    = (_Float16*)(ws + alloc((size_t)1152 * 1024 * 2));

  f32_to_f16<<<(1024 * 4480 + 255) / 256, 256, 0, stream>>>(W_in, Wi16, 1024 * 4480);
  f32_to_f16<<<(1152 * 1024 + 255) / 256, 256, 0, stream>>>(W_out, Wo16, 1152 * 1024);
  ln_kernel<<<16384, 256, 0, stream>>>(x, ln_w, ln_b, hln);
  conv_silu<<<65536, 256, 0, stream>>>(hln, conv_w, conv_b, h16);
  gemm_wmma<true, false><<<dim3(128, 70), 256, 0, stream>>>(h16, Wi16, b_in, nullptr,
                                                            nullptr, projs, 16384, 4480, 1024, 4480);
  ssm_scan<<<512, 32, 0, stream>>>(projs, A_log, concat);
  attn_intra<<<dim3(32, 16, 4), 256, 0, stream>>>(projs, gamma, yintra, kvbuf);
  chunk_states<<<64, 256, 0, stream>>>(kvbuf, gamma, statesH);
  attn_final<<<dim3(32, 16, 4), 256, 0, stream>>>(projs, statesH, yintra, gamma, gn_w, gn_b, concat);
  gemm_wmma<false, true><<<dim3(128, 16), 256, 0, stream>>>(concat, Wo16, b_out, x,
                                                            out, nullptr, 16384, 1024, 1152, 1024);
}
